// MoEMLP_82643760709756
// MI455X (gfx1250) — compile-verified, matrix operand order checked by reading, not run on hardware
//
#include <hip/hip_runtime.h>
#include <cstdint>
#include <cstddef>

// ---------------- problem constants ----------------
constexpr int T_   = 16384;      // B*S tokens
constexpr int D_   = 2048;
constexpr int E_   = 8;
constexpr int TOPK = 2;
constexpr int I_   = 1408;
constexpr int J_   = 5632;
constexpr int CAP  = 5120;       // ceil(T*K/E * 1.25)

// ---------------- GEMM tiling ----------------
constexpr int BM = 128, BN = 128, BK = 32;
constexpr int LDSK = 40;   // 80B row stride = 20 banks -> conflict-free ds_load_b128

typedef __attribute__((ext_vector_type(16))) __bf16 v16bf;
typedef __attribute__((ext_vector_type(8)))  float  v8f;

// ---- gfx1250 async global->LDS path (guarded; sync fallback otherwise) ----
#if defined(__has_builtin)
#  if __has_builtin(__builtin_amdgcn_global_load_async_to_lds_b128)
#    define USE_ASYNC 1
#  endif
#endif
#ifndef USE_ASYNC
#  define USE_ASYNC 0
#endif

#if USE_ASYNC
typedef int v4i_ __attribute__((vector_size(4 * sizeof(int))));
typedef __attribute__((address_space(1))) v4i_* gbl_v4i;
typedef __attribute__((address_space(3))) v4i_* lds_v4i;

static __device__ __forceinline__ void async_cp16(const void* g, void* lds) {
  // generic LDS ptr: low 32 bits are the LDS offset -> AS(3)
  __builtin_amdgcn_global_load_async_to_lds_b128(
      (gbl_v4i)(uintptr_t)g,
      (lds_v4i)(uint32_t)(uintptr_t)lds,
      0, 0);
}
static __device__ __forceinline__ void wait_async0() {
#  if __has_builtin(__builtin_amdgcn_s_wait_asynccnt)
  __builtin_amdgcn_s_wait_asynccnt(0);
#  else
  asm volatile("s_wait_asynccnt 0x0" ::: "memory");
#  endif
}
#endif

// fp32 -> bf16 bits, round-to-nearest-even
static __device__ __forceinline__ unsigned short f2bf(float f) {
  union { float f; uint32_t u; } v; v.f = f;
  uint32_t r = v.u + 0x7FFFu + ((v.u >> 16) & 1u);
  return (unsigned short)(r >> 16);
}

static __device__ __forceinline__ v8f wmma_bf16(v16bf a, v16bf b, v8f c) {
  // (neg_a, A, neg_b, B, c_mod, C, reuse_a, reuse_b)
  return __builtin_amdgcn_wmma_f32_16x16x32_bf16(false, a, false, b, (short)0, c, false, false);
}

// Load one 16x32 bf16 fragment from an LDS tile stored [row][k] with stride LDSK.
// 16-bit A layout: lanes 0-15 hold K=0..7 (v0-3) / K=16..23 (v4-7), lanes 16-31
// hold K=8..15 / K=24..31.  Two ds_load_b128 per lane.
static __device__ __forceinline__ v16bf ldfrag(const unsigned short* sm, int row0,
                                               int lr, int half) {
  const unsigned short* p = sm + (row0 + lr) * LDSK + half * 8;
  union { uint4 q[2]; v16bf v; } u;
  u.q[0] = *(const uint4*)p;
  u.q[1] = *(const uint4*)(p + 16);
  return u.v;
}

// ---------------- init: counters + slot map (must run every launch) ----------------
__global__ void init_meta(int* __restrict__ cnt, int* __restrict__ tok, int n) {
  int i = blockIdx.x * 256 + threadIdx.x;
  if (i < E_) cnt[i] = 0;
  if (i < n)  tok[i] = -1;
}

// ---------------- fp32 -> bf16, 4 elems/thread ----------------
__global__ void cvt_bf16(const float* __restrict__ in, unsigned short* __restrict__ out, int n4) {
  int i = blockIdx.x * blockDim.x + threadIdx.x;
  if (i >= n4) return;
  float4 v = ((const float4*)in)[i];
  union { unsigned short s[4]; uint2 u; } o;
  o.s[0] = f2bf(v.x); o.s[1] = f2bf(v.y); o.s[2] = f2bf(v.z); o.s[3] = f2bf(v.w);
  ((uint2*)out)[i] = o.u;
}

// ---------------- tiled transpose + convert: in[z][rows][cols] f32 -> out[z][cols][rows] bf16 ----
__global__ void tr_cvt(const float* __restrict__ in, unsigned short* __restrict__ out,
                       int rows, int cols) {
  __shared__ float tile[32][33];
  const size_t zofs = (size_t)blockIdx.z * rows * cols;
  const int c0 = blockIdx.x * 32, r0 = blockIdx.y * 32;
  const int tx = threadIdx.x, ty = threadIdx.y;   // (32,8)
  #pragma unroll
  for (int i = 0; i < 32; i += 8)
    tile[ty + i][tx] = in[zofs + (size_t)(r0 + ty + i) * cols + (c0 + tx)];
  __syncthreads();
  #pragma unroll
  for (int i = 0; i < 32; i += 8)
    out[zofs + (size_t)(c0 + ty + i) * rows + (r0 + tx)] = f2bf(tile[tx][ty + i]);
}

// ---------------- router + top2 + softmax: one wave per token ----------------
__global__ void router_topk(const float* __restrict__ x, const float* __restrict__ Wr,
                            int* __restrict__ sel, float* __restrict__ cw) {
  const int wave = threadIdx.x >> 5, lane = threadIdx.x & 31;
  const int t = blockIdx.x * 8 + wave;
  const float* xr = x + (size_t)t * D_;
  float acc[E_];
  #pragma unroll
  for (int e = 0; e < E_; ++e) acc[e] = 0.f;
  for (int d = lane; d < D_; d += 32) {
    const float xv = xr[d];
    const float* w = Wr + d * E_;
    #pragma unroll
    for (int e = 0; e < E_; ++e) acc[e] += xv * w[e];
  }
  #pragma unroll
  for (int off = 16; off > 0; off >>= 1)
    #pragma unroll
    for (int e = 0; e < E_; ++e) acc[e] += __shfl_xor(acc[e], off, 32);
  if (lane == 0) {
    float best = acc[0], second = -3.4e38f;
    int be = 0, se = 0;
    #pragma unroll
    for (int e = 1; e < E_; ++e) {
      if (acc[e] > best) { second = best; se = be; best = acc[e]; be = e; }
      else if (acc[e] > second) { second = acc[e]; se = e; }
    }
    const float w1 = __expf(second - best);
    const float inv = 1.f / (1.f + w1);
    sel[t * 2 + 0] = be;  sel[t * 2 + 1] = se;
    cw[t * 2 + 0] = inv;  cw[t * 2 + 1] = w1 * inv;
  }
}

// ---------------- capacity dispatch: one wave per (token, k) slot ----------------
__global__ void dispatch_kernel(const float* __restrict__ x, const int* __restrict__ sel,
                                const float* __restrict__ cw, int* __restrict__ cnt,
                                int* __restrict__ tok_of, float* __restrict__ w_of,
                                unsigned short* __restrict__ buf) {
  const int wave = threadIdx.x >> 5, lane = threadIdx.x & 31;
  const int s = blockIdx.x * 8 + wave;           // slot in [0, T*K)
  const int t = s >> 1;
  int e = 0, pos = 0;
  if (lane == 0) { e = sel[s]; pos = atomicAdd(cnt + e, 1); }
  e = __shfl(e, 0, 32); pos = __shfl(pos, 0, 32);
  if (pos >= CAP) return;                        // dropped token
  if (lane == 0) { tok_of[e * CAP + pos] = t; w_of[e * CAP + pos] = cw[s]; }
  const float4* src = (const float4*)(x + (size_t)t * D_);
  uint2* dst = (uint2*)(buf + ((size_t)e * CAP + pos) * D_);
  for (int c = lane; c < D_ / 4; c += 32) {
    float4 v = src[c];
    union { unsigned short s4[4]; uint2 u; } o;
    o.s4[0] = f2bf(v.x); o.s4[1] = f2bf(v.y); o.s4[2] = f2bf(v.z); o.s4[3] = f2bf(v.w);
    dst[c] = o.u;
  }
}

// ---------------- WMMA GEMM: C[M,N] = A[M,K] * Bt[N,K]^T ----------------
// MODE 0: dual-B SwiGLU epilogue -> bf16 h   (Bt gate rows [0,N), up rows [upOfs,upOfs+N))
// MODE 1: plain f32 store -> OutP
// MODE 2: scatter  atomicAdd(y[tok_of[row]], w_of[row] * val)
template <int MODE>
__global__ __launch_bounds__(256) void gemm_bf16_wmma(
    const unsigned short* __restrict__ A,  long long sA,
    const unsigned short* __restrict__ Bt, long long sB,
    void* __restrict__ OutP, long long sO,
    int M, int N, int Kd, int upOfs,
    const int* __restrict__ counters, int cap,
    const int* __restrict__ tok_of, const float* __restrict__ w_of,
    float* __restrict__ y, int yld) {
#if USE_ASYNC
  constexpr int NSTAGE = 2;          // double-buffered, async DMA overlaps WMMA
#else
  constexpr int NSTAGE = 1;
#endif
  __shared__ __align__(16) unsigned short smA[NSTAGE][BM * LDSK];
  __shared__ __align__(16) unsigned short smB[NSTAGE][BM * LDSK];
  __shared__ __align__(16) unsigned short smB2[MODE == 0 ? NSTAGE : 1]
                                              [MODE == 0 ? BM * LDSK : 16];

  const int z = blockIdx.z;
  const unsigned short* Ab = A + (size_t)z * sA;
  const unsigned short* Bb = Bt + (size_t)z * sB;
  const int m0 = blockIdx.y * BM, n0 = blockIdx.x * BN;
  if (counters) {                                  // uniform early-exit, EXEC stays all-ones
    int c = counters[z]; c = c < cap ? c : cap;
    if (m0 >= c) return;
  }
  const int tid = threadIdx.x;
  const int lane = tid & 31, wave = tid >> 5;
  const int wm = wave >> 2, wn = wave & 3;         // 2x4 wave grid, wave tile 64x32
  const int half = lane >> 4, lr = lane & 15;

  v8f acc[4][2] = {};
  v8f acc2[4][2] = {};

#if USE_ASYNC
  auto stage_issue = [&](int st, int k0) {
    for (int c = tid; c < (BM * BK) / 8; c += 256) {   // 512 16B chunks, 2/thread
      const int row = c >> 2, ko = (c & 3) << 3;
      async_cp16(&Ab[(size_t)(m0 + row) * Kd + (k0 + ko)], &smA[st][row * LDSK + ko]);
      async_cp16(&Bb[(size_t)(n0 + row) * Kd + (k0 + ko)], &smB[st][row * LDSK + ko]);
      if (MODE == 0)
        async_cp16(&Bb[(size_t)(upOfs + n0 + row) * Kd + (k0 + ko)],
                   &smB2[st][row * LDSK + ko]);
    }
  };
#endif

  auto compute = [&](int st) {
    v16bf bf0 = ldfrag(&smB[st][0], wn * 32, lr, half);
    v16bf bf1 = ldfrag(&smB[st][0], wn * 32 + 16, lr, half);
    v16bf bg0, bg1;
    if (MODE == 0) {
      bg0 = ldfrag(&smB2[st][0], wn * 32, lr, half);
      bg1 = ldfrag(&smB2[st][0], wn * 32 + 16, lr, half);
    }
    #pragma unroll
    for (int i = 0; i < 4; ++i) {
      v16bf af = ldfrag(&smA[st][0], wm * 64 + i * 16, lr, half);
      acc[i][0] = wmma_bf16(af, bf0, acc[i][0]);
      acc[i][1] = wmma_bf16(af, bf1, acc[i][1]);
      if (MODE == 0) {
        acc2[i][0] = wmma_bf16(af, bg0, acc2[i][0]);
        acc2[i][1] = wmma_bf16(af, bg1, acc2[i][1]);
      }
    }
  };

  const int nk = Kd / BK;
#if USE_ASYNC
  stage_issue(0, 0);
  for (int kt = 0; kt < nk; ++kt) {
    const int cur = kt & 1;
    wait_async0();            // own DMA done
    __syncthreads();          // everyone's DMA done + prev reads of other stage done
    if (kt + 1 < nk) stage_issue(1 - cur, (kt + 1) * BK);
    compute(cur);
  }
#else
  for (int kt = 0; kt < nk; ++kt) {
    const int k0 = kt * BK;
    __syncthreads();
    for (int c = tid; c < (BM * BK) / 8; c += 256) {
      const int row = c >> 2, ko = (c & 3) << 3;
      *(uint4*)&smA[0][row * LDSK + ko] = *(const uint4*)&Ab[(size_t)(m0 + row) * Kd + (k0 + ko)];
      *(uint4*)&smB[0][row * LDSK + ko] = *(const uint4*)&Bb[(size_t)(n0 + row) * Kd + (k0 + ko)];
      if (MODE == 0)
        *(uint4*)&smB2[0][row * LDSK + ko] =
            *(const uint4*)&Bb[(size_t)(upOfs + n0 + row) * Kd + (k0 + ko)];
    }
    if (k0 + BK < Kd) {
      __builtin_prefetch(&Ab[(size_t)(m0 + (tid & 127)) * Kd + (k0 + BK)], 0, 1);
      __builtin_prefetch(&Bb[(size_t)(n0 + (tid & 127)) * Kd + (k0 + BK)], 0, 1);
    }
    __syncthreads();
    compute(0);
  }
#endif

  // epilogue.  C/D layout: VGPR r -> row (r + 8*half), col = lane%16.
  #pragma unroll
  for (int i = 0; i < 4; ++i)
    #pragma unroll
    for (int j = 0; j < 2; ++j)
      #pragma unroll
      for (int r = 0; r < 8; ++r) {
        const int grow = m0 + wm * 64 + i * 16 + half * 8 + r;
        const int gcol = n0 + wn * 32 + j * 16 + lr;
        const float v = acc[i][j][r];
        if (MODE == 0) {
          const float u = acc2[i][j][r];
          const float hh = v * (1.f / (1.f + __expf(-v))) * u;   // silu(gate)*up
          ((unsigned short*)OutP)[(size_t)z * sO + (size_t)grow * N + gcol] = f2bf(hh);
        } else if (MODE == 1) {
          ((float*)OutP)[(size_t)grow * N + gcol] = v;
        } else {
          const int tk = tok_of[(size_t)z * M + grow];
          if (tk >= 0)
            atomicAdd(&y[(size_t)tk * yld + gcol], w_of[(size_t)z * M + grow] * v);
        }
      }
}

// ---------------- host launcher ----------------
extern "C" void kernel_launch(void* const* d_in, const int* in_sizes, int n_in,
                              void* d_out, int out_size, void* d_ws, size_t ws_size,
                              hipStream_t stream) {
  const float* x    = (const float*)d_in[0];
  const float* Wr   = (const float*)d_in[1];
  const float* Wug  = (const float*)d_in[2];   // [E][D][2I]
  const float* Wd   = (const float*)d_in[3];   // [E][I][D]
  const float* Wsug = (const float*)d_in[4];   // [D][2J]
  const float* Wsd  = (const float*)d_in[5];   // [J][D]
  float* y = (float*)d_out;

  char* ws = (char*)d_ws;
  size_t off = 0;
  auto alloc = [&](size_t bytes) { size_t o = off; off += (bytes + 255) & ~(size_t)255; return o; };
  const size_t o_xb    = alloc((size_t)T_ * D_ * 2);           // x bf16
  const size_t o_wugT  = alloc((size_t)E_ * 2 * I_ * D_ * 2);  // [E][2I][D]
  const size_t o_wdT   = alloc((size_t)E_ * D_ * I_ * 2);      // [E][D][I]
  const size_t o_wsugT = alloc((size_t)2 * J_ * D_ * 2);       // [2J][D]
  const size_t o_wsdT  = alloc((size_t)D_ * J_ * 2);           // [D][J]
  const size_t o_buf   = alloc((size_t)E_ * CAP * D_ * 2);     // dispatched tokens bf16
  const size_t o_h     = alloc((size_t)E_ * CAP * I_ * 2);     // routed hidden bf16
  const size_t o_hs    = alloc((size_t)T_ * J_ * 2);           // shared hidden bf16
  const size_t o_sel   = alloc((size_t)T_ * TOPK * 4);
  const size_t o_cw    = alloc((size_t)T_ * TOPK * 4);
  const size_t o_cnt   = alloc(256);
  const size_t o_tok   = alloc((size_t)E_ * CAP * 4);
  const size_t o_w     = alloc((size_t)E_ * CAP * 4);

  unsigned short* xb    = (unsigned short*)(ws + o_xb);
  unsigned short* wugT  = (unsigned short*)(ws + o_wugT);
  unsigned short* wdT   = (unsigned short*)(ws + o_wdT);
  unsigned short* wsugT = (unsigned short*)(ws + o_wsugT);
  unsigned short* wsdT  = (unsigned short*)(ws + o_wsdT);
  unsigned short* buf   = (unsigned short*)(ws + o_buf);
  unsigned short* h     = (unsigned short*)(ws + o_h);
  unsigned short* hs    = (unsigned short*)(ws + o_hs);
  int*   sel = (int*)(ws + o_sel);
  float* cw  = (float*)(ws + o_cw);
  int*   cnt = (int*)(ws + o_cnt);
  int*   tok = (int*)(ws + o_tok);
  float* wgt = (float*)(ws + o_w);

  // 1. reset per-call state
  init_meta<<<(E_ * CAP + 255) / 256, 256, 0, stream>>>(cnt, tok, E_ * CAP);
  // 2. converts / transposes
  cvt_bf16<<<(T_ * D_ / 4 + 255) / 256, 256, 0, stream>>>(x, xb, T_ * D_ / 4);
  tr_cvt<<<dim3(2 * I_ / 32, D_ / 32, E_), dim3(32, 8), 0, stream>>>(Wug, wugT, D_, 2 * I_);
  tr_cvt<<<dim3(D_ / 32, I_ / 32, E_), dim3(32, 8), 0, stream>>>(Wd, wdT, I_, D_);
  tr_cvt<<<dim3(2 * J_ / 32, D_ / 32, 1), dim3(32, 8), 0, stream>>>(Wsug, wsugT, D_, 2 * J_);
  tr_cvt<<<dim3(D_ / 32, J_ / 32, 1), dim3(32, 8), 0, stream>>>(Wsd, wsdT, J_, D_);
  // 3. router + dispatch
  router_topk<<<T_ / 8, 256, 0, stream>>>(x, Wr, sel, cw);
  dispatch_kernel<<<(T_ * TOPK) / 8, 256, 0, stream>>>(x, sel, cw, cnt, tok, wgt, buf);
  // 4. routed up+gate (SwiGLU fused):  h[e] = silu(buf*Wg) * (buf*Wu)
  gemm_bf16_wmma<0><<<dim3(I_ / BN, CAP / BM, E_), 256, 0, stream>>>(
      buf, (long long)CAP * D_, wugT, (long long)2 * I_ * D_,
      h, (long long)CAP * I_, CAP, I_, D_, I_, cnt, CAP, nullptr, nullptr, nullptr, 0);
  // 5. shared up+gate (SwiGLU fused):  hs = silu(xb*Wsg) * (xb*Wsu)
  gemm_bf16_wmma<0><<<dim3(J_ / BN, T_ / BM, 1), 256, 0, stream>>>(
      xb, 0, wsugT, 0, hs, 0, T_, J_, D_, J_, nullptr, 0, nullptr, nullptr, nullptr, 0);
  // 6. shared down -> writes every d_out element (covers poison)
  gemm_bf16_wmma<1><<<dim3(D_ / BN, T_ / BM, 1), 256, 0, stream>>>(
      hs, 0, wsdT, 0, y, 0, T_, D_, J_, 0, nullptr, 0, nullptr, nullptr, nullptr, 0);
  // 7. routed down -> weighted scatter-add into d_out (after shared store)
  gemm_bf16_wmma<2><<<dim3(D_ / BN, CAP / BM, E_), 256, 0, stream>>>(
      h, (long long)CAP * I_, wdT, (long long)D_ * I_,
      nullptr, 0, CAP, D_, I_, 0, cnt, CAP, tok, wgt, y, D_);
  (void)in_sizes; (void)n_in; (void)out_size; (void)ws_size;
}
